// AdaATIN_66236985639011
// MI455X (gfx1250) — compile-verified, matrix operand order checked by reading, not run on hardware
//
#include <hip/hip_runtime.h>
#include <hip/hip_bf16.h>

// ---------------------------------------------------------------------------
// Problem constants
// ---------------------------------------------------------------------------
#define Bn 4
#define Cn 256
#define Hn 96
#define Wn 96
#define NHn 128
#define HWn (Hn * Wn)          // 9216
#define NPIX (Bn * HWn)        // 36864
#define PI_F 3.14159f

typedef __attribute__((ext_vector_type(16))) _Float16 v16h;
typedef __attribute__((ext_vector_type(8)))  float    v8f;

// ---------------------------------------------------------------------------
// Workspace layout (bytes)
// ---------------------------------------------------------------------------
static constexpr size_t OFF_WPAR  = 0;                      // [B*C][5] f32
static constexpr size_t OFF_G1    = 20480;                  // [B*C] f32
static constexpr size_t OFF_B1    = 24576;
static constexpr size_t OFF_G2    = 28672;
static constexpr size_t OFF_B2    = 32768;
static constexpr size_t OFF_STATS = 36864;                  // [B*C][2] f32
static constexpr size_t OFF_BUFA  = 45056;                  // NCHW f32 warped
static constexpr size_t OFF_ACT   = OFF_BUFA + 37748736;    // NHWC f16 act
static constexpr size_t OFF_WP1   = OFF_ACT + 18874368;     // [9][O][I] f16
static constexpr size_t OFF_WP2   = OFF_WP1 + 1179648;

// ---------------------------------------------------------------------------
// Kernel 0: all the small MLPs (para -> warp params, style -> gamma/beta)
// ---------------------------------------------------------------------------
__global__ __launch_bounds__(256) void prep_params(
    const float* __restrict__ para, const float* __restrict__ style,
    const float* __restrict__ cw,  const float* __restrict__ cb,
    const float* __restrict__ sw,  const float* __restrict__ sb,
    const float* __restrict__ rw,  const float* __restrict__ rb,
    const float* __restrict__ tw,  const float* __restrict__ tb,
    const float* __restrict__ a1sw, const float* __restrict__ a1sb,
    const float* __restrict__ a1gw, const float* __restrict__ a1gb,
    const float* __restrict__ a1bw, const float* __restrict__ a1bb,
    const float* __restrict__ a2sw, const float* __restrict__ a2sb,
    const float* __restrict__ a2gw, const float* __restrict__ a2gb,
    const float* __restrict__ a2bw, const float* __restrict__ a2bb,
    float* __restrict__ wparams,
    float* __restrict__ g1, float* __restrict__ b1,
    float* __restrict__ g2, float* __restrict__ b2)
{
    __shared__ float pcr[Bn][256];
    __shared__ float ah1[Bn][NHn];
    __shared__ float ah2[Bn][NHn];
    const int t = threadIdx.x;

    for (int b = 0; b < Bn; ++b) {
        float acc = cb[t];
        for (int k = 0; k < 256; ++k)
            acc = fmaf(para[b * 256 + k], cw[k * 256 + t], acc);
        pcr[b][t] = fmaxf(acc, 0.f);
    }
    for (int i = t; i < Bn * NHn; i += 256) {
        const int b = i >> 7, h = i & (NHn - 1);
        float a1 = a1sb[h], a2 = a2sb[h];
        for (int k = 0; k < 256; ++k) {
            const float s = style[b * 256 + k];
            a1 = fmaf(s, a1sw[k * NHn + h], a1);
            a2 = fmaf(s, a2sw[k * NHn + h], a2);
        }
        ah1[b][h] = fmaxf(a1, 0.f);
        ah2[b][h] = fmaxf(a2, 0.f);
    }
    __syncthreads();

    for (int i = t; i < Bn * Cn; i += 256) {
        const int b = i >> 8, c = i & 255;
        float ds = sb[c], dr = rb[c], dtx = tb[2 * c], dty = tb[2 * c + 1];
        for (int k = 0; k < 256; ++k) {
            const float p = pcr[b][k];
            ds  = fmaf(p, sw[k * 256 + c], ds);
            dr  = fmaf(p, rw[k * 256 + c], dr);
            dtx = fmaf(p, tw[k * 512 + 2 * c],     dtx);
            dty = fmaf(p, tw[k * 512 + 2 * c + 1], dty);
        }
        const float scale = 2.f / (1.f + expf(-ds));
        const float ang   = tanhf(dr) * PI_F;
        float* wp = wparams + (size_t)i * 5;
        wp[0] = scale;
        wp[1] = cosf(ang);
        wp[2] = sinf(ang);
        wp[3] = tanhf(dtx);
        wp[4] = tanhf(dty);

        float ga1 = a1gb[c], be1 = a1bb[c], ga2 = a2gb[c], be2 = a2bb[c];
        for (int h = 0; h < NHn; ++h) {
            const float v1 = ah1[b][h], v2 = ah2[b][h];
            ga1 = fmaf(v1, a1gw[h * 256 + c], ga1);
            be1 = fmaf(v1, a1bw[h * 256 + c], be1);
            ga2 = fmaf(v2, a2gw[h * 256 + c], ga2);
            be2 = fmaf(v2, a2bw[h * 256 + c], be2);
        }
        g1[i] = ga1; b1[i] = be1; g2[i] = ga2; b2[i] = be2;
    }
}

// ---------------------------------------------------------------------------
// Kernel 1: pack conv weights OIHW f32 -> [kk][O][I] f16
// ---------------------------------------------------------------------------
__global__ __launch_bounds__(256) void pack_weights(
    const float* __restrict__ w, _Float16* __restrict__ wp)
{
    const int idx = blockIdx.x * 256 + threadIdx.x;   // 9*256*256 total
    if (idx >= 9 * 256 * 256) return;
    const int ci = idx & 255;
    const int n  = (idx >> 8) & 255;
    const int kk = idx >> 16;
    wp[idx] = (_Float16)w[(n * 256 + ci) * 9 + kk];
}

// ---------------------------------------------------------------------------
// Kernel 2: AdaAT trilinear warp
// ---------------------------------------------------------------------------
__global__ __launch_bounds__(256) void warp_kernel(
    const float* __restrict__ fm, const float* __restrict__ wparams,
    float* __restrict__ outp)
{
    const int idx = blockIdx.x * 256 + threadIdx.x;   // NCHW order
    const int x = idx % Wn;
    const int y = (idx / Wn) % Hn;
    const int c = (idx / HWn) % Cn;
    const int b = idx / (Cn * HWn);

    const float* p = wparams + (size_t)(b * Cn + c) * 5;
    const float s = p[0], cs = p[1], sn = p[2], tx = p[3], ty = p[4];

    const float gx = 2.f * x / (Wn - 1) - 1.f;
    const float gy = 2.f * y / (Hn - 1) - 1.f;
    const float xw = (cs * gx - sn * gy) * s + tx;
    const float yw = (sn * gx + cs * gy) * s + ty;
    const float ix = ((xw + 1.f) * Wn - 1.f) * 0.5f;
    const float iy = ((yw + 1.f) * Hn - 1.f) * 0.5f;
    const float gz = 2.f * c / (Cn - 1) - 1.f;
    const float iz = ((gz + 1.f) * Cn - 1.f) * 0.5f;

    const float x0f = floorf(ix), y0f = floorf(iy), z0f = floorf(iz);
    const float wx = ix - x0f, wy = iy - y0f, wz = iz - z0f;
    const int x0 = (int)x0f, y0 = (int)y0f, z0 = (int)z0f;

    const float* fb = fm + (size_t)b * Cn * HWn;
    float acc = 0.f;
#pragma unroll
    for (int dz = 0; dz < 2; ++dz) {
        const int zi = z0 + dz;
        const float wzc = dz ? wz : 1.f - wz;
        const bool zv = (zi >= 0) & (zi < Cn);
        const int zc = min(max(zi, 0), Cn - 1);
#pragma unroll
        for (int dy = 0; dy < 2; ++dy) {
            const int yi = y0 + dy;
            const float wyc = dy ? wy : 1.f - wy;
            const bool yv = (yi >= 0) & (yi < Hn);
            const int yc = min(max(yi, 0), Hn - 1);
#pragma unroll
            for (int dx = 0; dx < 2; ++dx) {
                const int xi = x0 + dx;
                const float wxc = dx ? wx : 1.f - wx;
                const bool v = zv & yv & (xi >= 0) & (xi < Wn);
                const int xc = min(max(xi, 0), Wn - 1);
                const float val = v ? fb[((size_t)zc * Hn + yc) * Wn + xc] : 0.f;
                acc = fmaf(val, wzc * wyc * wxc, acc);
            }
        }
    }
    outp[idx] = acc;
}

// ---------------------------------------------------------------------------
// Kernel 3: per-(b,c) mean / rstd over H*W
// ---------------------------------------------------------------------------
__global__ __launch_bounds__(256) void stats_kernel(
    const float* __restrict__ xin, float* __restrict__ stats)
{
    __shared__ float ssum[256], ssq[256];
    const int bc = blockIdx.x;
    const float* p = xin + (size_t)bc * HWn;
    float s = 0.f, q = 0.f;
    for (int i = threadIdx.x; i < HWn; i += 256) {
        const float v = p[i];
        s += v; q = fmaf(v, v, q);
    }
    ssum[threadIdx.x] = s; ssq[threadIdx.x] = q;
    __syncthreads();
    for (int off = 128; off > 0; off >>= 1) {
        if (threadIdx.x < off) {
            ssum[threadIdx.x] += ssum[threadIdx.x + off];
            ssq[threadIdx.x]  += ssq[threadIdx.x + off];
        }
        __syncthreads();
    }
    if (threadIdx.x == 0) {
        const float m = ssum[0] * (1.f / HWn);
        const float var = ssq[0] * (1.f / HWn) - m * m;
        stats[bc * 2 + 0] = m;
        stats[bc * 2 + 1] = rsqrtf(var + 1e-5f);
    }
}

// ---------------------------------------------------------------------------
// Kernel 4: AdaIN apply + LeakyReLU + f32 NCHW -> f16 NHWC
// ---------------------------------------------------------------------------
__global__ __launch_bounds__(256) void normalize_kernel(
    const float* __restrict__ xin, const float* __restrict__ stats,
    const float* __restrict__ gamma, const float* __restrict__ beta,
    _Float16* __restrict__ act)
{
    const int idx = blockIdx.x * 256 + threadIdx.x;   // NHWC order, C innermost
    const int c = idx & 255;
    const int pix = idx >> 8;
    const int x = pix % Wn;
    const int y = (pix / Wn) % Hn;
    const int b = pix / HWn;
    const int bc = b * Cn + c;

    const float v = xin[((size_t)bc * Hn + y) * Wn + x];
    const float m = stats[bc * 2 + 0];
    const float r = stats[bc * 2 + 1];
    float o = (v - m) * r * (1.f + gamma[bc]) + beta[bc];
    o = o >= 0.f ? o : 0.2f * o;
    act[idx] = (_Float16)o;
}

// ---------------------------------------------------------------------------
// Kernel 5: implicit-GEMM 3x3 conv via v_wmma_f32_16x16x32_f16
//   wave tile  = 32 pixels x 32 out-channels (4 accumulators, 4 WMMA/K-step)
//   block tile = 128 pixels x 64 out-channels (8 waves: 4M x 2N)
//   B (weight) tiles double-buffered in LDS via global_load_async_to_lds_b128,
//   synchronized with s_wait_asynccnt + workgroup barrier.
// ---------------------------------------------------------------------------
__global__ __launch_bounds__(256) void conv_wmma(
    const _Float16* __restrict__ act, const _Float16* __restrict__ wp,
    const float* __restrict__ bias, float* __restrict__ out)
{
    __shared__ _Float16 lds_b[2][64 * 32];   // 2 x 4KB weight tiles

    const int tid   = threadIdx.x;
    const int lane  = tid & 31;
    const int wave  = tid >> 5;
    const int waveM = wave & 3;                    // 4 M-waves
    const int waveN = wave >> 2;                   // 2 N-waves
    const int p0    = blockIdx.x * 128 + waveM * 32;
    const int nb0   = blockIdx.y * 64;

    const int lr = lane & 15;   // row/col within 16x16 fragment
    const int hi = lane >> 4;   // K offset select {0,8} per ISA layout

    // A-row pixel coordinates for the two M-subtiles
    int bA[2], yA[2], xA[2];
#pragma unroll
    for (int m = 0; m < 2; ++m) {
        const int p = p0 + m * 16 + lr;
        bA[m] = p / HWn;
        const int rem = p - bA[m] * HWn;
        yA[m] = rem / Wn;
        xA[m] = rem - yA[m] * Wn;
    }

    // Per-thread staging slot: 256 threads x 16B == 64ch x 32 halves tile
    const int      nlS = tid >> 2;                 // 0..63 (channel in tile)
    const int      cpS = (tid & 3) << 3;           // 0,8,16,24 (halves)
    const unsigned ldsBase = (unsigned)(size_t)(&lds_b[0][0]);
    const unsigned ldsSlot = (unsigned)((nlS * 32 + cpS) * 2);

    // Issue async copy of K-step s's weight tile into LDS buffer (s&1)
    auto issueB = [&](int s) {
        const int kk = s >> 3;
        const int cb = (s & 7) << 5;
        const _Float16* src =
            wp + ((size_t)(kk * Cn + nb0 + nlS)) * Cn + cb + cpS;
        const unsigned dst = ldsBase + (unsigned)((s & 1) * 4096) + ldsSlot;
        asm volatile("global_load_async_to_lds_b128 %0, %1, off"
                     :: "v"(dst), "v"(src) : "memory");
    };

    v8f acc[2][2] = {};

    issueB(0);
    asm volatile("s_wait_asynccnt 0" ::: "memory");
    __syncthreads();

    for (int s = 0; s < 72; ++s) {
        if (s + 1 < 72) issueB(s + 1);

        const int kk = s >> 3;
        const int ky = (kk * 11) >> 5;             // kk/3 for kk in [0,9)
        const int kx = kk - ky * 3;
        const int cb = (s & 7) << 5;

        // B fragments from LDS (ds_load_b128 pairs)
        const _Float16* lb = &lds_b[s & 1][0];
        v16h bfr[2];
#pragma unroll
        for (int nSub = 0; nSub < 2; ++nSub) {
            const _Float16* bp =
                lb + (waveN * 32 + nSub * 16 + lr) * 32 + hi * 8;
            v16h f;
            ((uint4*)&f)[0] = *(const uint4*)(bp);
            ((uint4*)&f)[1] = *(const uint4*)(bp + 16);
            bfr[nSub] = f;
        }

        // A fragments from global (NHWC, contiguous in C) + 4 WMMAs
#pragma unroll
        for (int mSub = 0; mSub < 2; ++mSub) {
            const int ys = yA[mSub] + ky - 1;
            const int xs = xA[mSub] + kx - 1;
            const bool valid = (ys >= 0) & (ys < Hn) & (xs >= 0) & (xs < Wn);
            const _Float16* arow =
                act + ((size_t)(bA[mSub] * Hn + ys) * Wn + xs) * Cn + hi * 8;
            uint4 a0 = {0u, 0u, 0u, 0u}, a1 = {0u, 0u, 0u, 0u};
            if (valid) {
                a0 = *(const uint4*)(arow + cb);
                a1 = *(const uint4*)(arow + cb + 16);
            }
            v16h af;
            ((uint4*)&af)[0] = a0;
            ((uint4*)&af)[1] = a1;
            acc[mSub][0] = __builtin_amdgcn_wmma_f32_16x16x32_f16(
                false, af, false, bfr[0], (short)0, acc[mSub][0], false, false);
            acc[mSub][1] = __builtin_amdgcn_wmma_f32_16x16x32_f16(
                false, af, false, bfr[1], (short)0, acc[mSub][1], false, false);
        }

        if (s + 1 < 72) {
            asm volatile("s_wait_asynccnt 0" ::: "memory");
            __syncthreads();
        }
    }

    // D layout: VGPR r -> row (r + hi*8), col lr. Write NCHW f32 + bias.
#pragma unroll
    for (int mSub = 0; mSub < 2; ++mSub) {
#pragma unroll
        for (int nSub = 0; nSub < 2; ++nSub) {
            const int nout = nb0 + waveN * 32 + nSub * 16 + lr;
            const float bs = bias[nout];
#pragma unroll
            for (int r = 0; r < 8; ++r) {
                const int pm = p0 + mSub * 16 + r + hi * 8;
                const int bb = pm / HWn;
                const int rm = pm - bb * HWn;
                out[((size_t)bb * Cn + nout) * HWn + rm] =
                    acc[mSub][nSub][r] + bs;
            }
        }
    }
}

// ---------------------------------------------------------------------------
// Host launcher
// ---------------------------------------------------------------------------
extern "C" void kernel_launch(void* const* d_in, const int* in_sizes, int n_in,
                              void* d_out, int out_size, void* d_ws, size_t ws_size,
                              hipStream_t stream) {
    (void)in_sizes; (void)n_in; (void)out_size; (void)ws_size;

    const float* fm        = (const float*)d_in[0];
    const float* para      = (const float*)d_in[1];
    const float* style     = (const float*)d_in[2];
    const float* cw        = (const float*)d_in[3];
    const float* cbb       = (const float*)d_in[4];
    const float* sw        = (const float*)d_in[5];
    const float* sbb       = (const float*)d_in[6];
    const float* rw        = (const float*)d_in[7];
    const float* rbb       = (const float*)d_in[8];
    const float* tw        = (const float*)d_in[9];
    const float* tbb       = (const float*)d_in[10];
    const float* a1sw      = (const float*)d_in[11];
    const float* a1sb      = (const float*)d_in[12];
    const float* a1gw      = (const float*)d_in[13];
    const float* a1gb      = (const float*)d_in[14];
    const float* a1bw      = (const float*)d_in[15];
    const float* a1bb      = (const float*)d_in[16];
    const float* conv1_w   = (const float*)d_in[17];
    const float* conv1_b   = (const float*)d_in[18];
    const float* a2sw      = (const float*)d_in[19];
    const float* a2sb      = (const float*)d_in[20];
    const float* a2gw      = (const float*)d_in[21];
    const float* a2gb      = (const float*)d_in[22];
    const float* a2bw      = (const float*)d_in[23];
    const float* a2bb      = (const float*)d_in[24];
    const float* conv2_w   = (const float*)d_in[25];
    const float* conv2_b   = (const float*)d_in[26];

    char* ws = (char*)d_ws;
    float*     wpar  = (float*)(ws + OFF_WPAR);
    float*     g1    = (float*)(ws + OFF_G1);
    float*     b1    = (float*)(ws + OFF_B1);
    float*     g2    = (float*)(ws + OFF_G2);
    float*     b2    = (float*)(ws + OFF_B2);
    float*     stats = (float*)(ws + OFF_STATS);
    float*     bufA  = (float*)(ws + OFF_BUFA);
    _Float16*  actb  = (_Float16*)(ws + OFF_ACT);
    _Float16*  wp1   = (_Float16*)(ws + OFF_WP1);
    _Float16*  wp2   = (_Float16*)(ws + OFF_WP2);
    float*     out   = (float*)d_out;

    prep_params<<<1, 256, 0, stream>>>(para, style, cw, cbb, sw, sbb, rw, rbb,
                                       tw, tbb, a1sw, a1sb, a1gw, a1gb, a1bw,
                                       a1bb, a2sw, a2sb, a2gw, a2gb, a2bw, a2bb,
                                       wpar, g1, b1, g2, b2);
    pack_weights<<<(9 * 256 * 256 + 255) / 256, 256, 0, stream>>>(conv1_w, wp1);
    pack_weights<<<(9 * 256 * 256 + 255) / 256, 256, 0, stream>>>(conv2_w, wp2);

    warp_kernel<<<(Bn * Cn * HWn) / 256, 256, 0, stream>>>(fm, wpar, bufA);

    stats_kernel<<<Bn * Cn, 256, 0, stream>>>(bufA, stats);
    normalize_kernel<<<(Bn * HWn * Cn) / 256, 256, 0, stream>>>(bufA, stats, g1, b1, actb);

    conv_wmma<<<dim3(NPIX / 128, Cn / 64), 256, 0, stream>>>(actb, wp1, conv1_b, out);

    stats_kernel<<<Bn * Cn, 256, 0, stream>>>(out, stats);
    normalize_kernel<<<(Bn * HWn * Cn) / 256, 256, 0, stream>>>(out, stats, g2, b2, actb);

    conv_wmma<<<dim3(NPIX / 128, Cn / 64), 256, 0, stream>>>(actb, wp2, conv2_b, out);
}
